// Seq2Seq_44942537786012
// MI455X (gfx1250) — matrix operand order repaired and checked by reading, not compiled, not run on hardware
//
#include <hip/hip_runtime.h>

// ---------------------------------------------------------------------------
// CDNA5 (gfx1250) seq2seq forward: bf16 WMMA (f32 accum) for all matmuls.
// Recurrent state h is carried in bf16 (c stays f32), so the serialized
// LSTM-step kernels load A fragments as raw b128 pairs with zero conversion
// ALU on the critical path. Wide parallel GEMMs (zx, K/V) keep f32 A with
// on-the-fly v_cvt_pk_bf16_f32 (co-executes under WMMA).
// All K dims padded to multiples of 32; all B (N) rows padded to tiles.
// ---------------------------------------------------------------------------

typedef __bf16 bf16_t;
typedef __attribute__((ext_vector_type(16))) __bf16 v16bf;
typedef __attribute__((ext_vector_type(8)))  __bf16 v8bf;
typedef __attribute__((ext_vector_type(8)))  float  v8f;

enum { ACT_NONE = 0, ACT_SELU = 1 };

__device__ __forceinline__ float sigm_f(float x) { return 1.0f / (1.0f + __expf(-x)); }
__device__ __forceinline__ float tanh_f(float x) {
  x = fminf(fmaxf(x, -15.0f), 15.0f);
  float e = __expf(2.0f * x);
  return (e - 1.0f) / (e + 1.0f);
}
__device__ __forceinline__ float selu_f(float x) {
  const float sc = 1.0507009873554805f, al = 1.6732632423543772f;
  return x > 0.0f ? sc * x : sc * al * (__expf(x) - 1.0f);
}

__device__ __forceinline__ v8f v8f_zero() {
  v8f z;
#pragma unroll
  for (int i = 0; i < 8; ++i) z[i] = 0.0f;
  return z;
}

// A-matrix fragment (16x32 bf16, MxK) from f32 row (convert on the fly).
// lane half=0: K = k0 + {0..7, 16..23}; half=1: K = k0 + {8..15, 24..31}.
__device__ __forceinline__ v16bf frag_a_f32(const float* __restrict__ row,
                                            int k0, int half) {
  const float* p0 = row + k0 + half * 8;
  float4 f0 = *(const float4*)(p0);
  float4 f1 = *(const float4*)(p0 + 4);
  float4 f2 = *(const float4*)(p0 + 16);
  float4 f3 = *(const float4*)(p0 + 20);
  v16bf a;
  a[0]  = (__bf16)f0.x; a[1]  = (__bf16)f0.y; a[2]  = (__bf16)f0.z; a[3]  = (__bf16)f0.w;
  a[4]  = (__bf16)f1.x; a[5]  = (__bf16)f1.y; a[6]  = (__bf16)f1.z; a[7]  = (__bf16)f1.w;
  a[8]  = (__bf16)f2.x; a[9]  = (__bf16)f2.y; a[10] = (__bf16)f2.z; a[11] = (__bf16)f2.w;
  a[12] = (__bf16)f3.x; a[13] = (__bf16)f3.y; a[14] = (__bf16)f3.z; a[15] = (__bf16)f3.w;
  return a;
}

// A-matrix fragment from a bf16 row: two aligned 16B loads, no conversion.
__device__ __forceinline__ v16bf frag_a_bf16(const bf16_t* __restrict__ row,
                                             int k0, int half) {
  const bf16_t* p = row + k0 + half * 8;
  v8bf lo = *(const v8bf*)(p);
  v8bf hi = *(const v8bf*)(p + 16);
  v16bf a;
#pragma unroll
  for (int e = 0; e < 8; ++e) { a[e] = lo[e]; a[8 + e] = hi[e]; }
  return a;
}

// B-matrix fragment (32x16 bf16, KxN) from bf16 weight row [N x K] row-major.
// lane half=0: K = k0 + {0..7, 16..23}; half=1: K = k0 + {8..15, 24..31}.
__device__ __forceinline__ v16bf frag_b_bf16(const bf16_t* __restrict__ row,
                                             int k0, int half) {
  const bf16_t* p = row + k0 + half * 8;
  v8bf lo = *(const v8bf*)(p);
  v8bf hi = *(const v8bf*)(p + 16);
  v16bf b;
#pragma unroll
  for (int e = 0; e < 8; ++e) { b[e] = lo[e]; b[8 + e] = hi[e]; }
  return b;
}

__device__ __forceinline__ v8f wmma_bf16(v16bf a, v16bf b, v8f c) {
  return __builtin_amdgcn_wmma_f32_16x16x32_bf16(false, a, false, b, (short)0, c,
                                                 false, false);
}

// ---------------------------------------------------------------------------
// GEMM (f32 A): C[M,N] = act(A[M,K] * B[N,K]^T + bias[N])
// block = 256 (8 waves); wave w -> tile (blockIdx.y*16, blockIdx.x*128+w*16).
// REQUIRES: M mult of 16, K mult of 32, B padded to >= tile rows.
// ---------------------------------------------------------------------------
__global__ __launch_bounds__(256) void gemm_bias_act(
    const float* __restrict__ A, const bf16_t* __restrict__ B,
    const float* __restrict__ bias, float* __restrict__ C,
    int M, int N, int K, int act) {
  int lane = threadIdx.x & 31, wave = threadIdx.x >> 5;
  int half = lane >> 4, r = lane & 15;
  int mo = blockIdx.y * 16;
  int no = blockIdx.x * 128 + wave * 16;
  if (no >= N) return;  // wave-uniform: EXEC stays all-ones for WMMA
  const float*  ap = A + (size_t)(mo + r) * K;
  const bf16_t* bp = B + (size_t)(no + r) * K;
  v8f acc = v8f_zero();
  for (int k0 = 0; k0 < K; k0 += 32) {
    v16bf a = frag_a_f32(ap, k0, half);
    v16bf b = frag_b_bf16(bp, k0, half);
    acc = wmma_bf16(a, b, acc);
  }
  int n = no + r;
  if (n < N) {
    float bb = (bias != nullptr) ? bias[n] : 0.0f;
#pragma unroll
    for (int v = 0; v < 8; ++v) {
      int m = mo + half * 8 + v;
      float x = acc[v] + bb;
      if (act == ACT_SELU) x = selu_f(x);
      C[(size_t)m * N + n] = x;
    }
  }
}

// Same GEMM but with a bf16 A operand (decoder q / logits path).
__global__ __launch_bounds__(256) void gemm_bf16A(
    const bf16_t* __restrict__ A, const bf16_t* __restrict__ B,
    const float* __restrict__ bias, float* __restrict__ C,
    int M, int N, int K, int act) {
  int lane = threadIdx.x & 31, wave = threadIdx.x >> 5;
  int half = lane >> 4, r = lane & 15;
  int mo = blockIdx.y * 16;
  int no = blockIdx.x * 128 + wave * 16;
  if (no >= N) return;
  const bf16_t* ap = A + (size_t)(mo + r) * K;
  const bf16_t* bp = B + (size_t)(no + r) * K;
  v8f acc = v8f_zero();
  for (int k0 = 0; k0 < K; k0 += 32) {
    v16bf a = frag_a_bf16(ap, k0, half);
    v16bf b = frag_b_bf16(bp, k0, half);
    acc = wmma_bf16(a, b, acc);
  }
  int n = no + r;
  if (n < N) {
    float bb = (bias != nullptr) ? bias[n] : 0.0f;
#pragma unroll
    for (int v = 0; v < 8; ++v) {
      int m = mo + half * 8 + v;
      float x = acc[v] + bb;
      if (act == ACT_SELU) x = selu_f(x);
      C[(size_t)m * N + n] = x;
    }
  }
}

// ---------------------------------------------------------------------------
// Encoder recurrent step, both directions in one launch (grid = {H/64, 2}).
// z = zx(precomputed x*Wih^T+b) + h*Whh^T ; gates fused in registers.
// h state carried in bf16 (ping-pong); c state f32 in-place; seq output f32.
// ---------------------------------------------------------------------------
__global__ __launch_bounds__(128) void lstm_step_enc(
    const float* __restrict__ zxf, const float* __restrict__ zxr,
    const bf16_t* __restrict__ Whf, const bf16_t* __restrict__ Whr,
    const bf16_t* __restrict__ hf_in, const bf16_t* __restrict__ hr_in,
    bf16_t* __restrict__ hf_out, bf16_t* __restrict__ hr_out,
    float* __restrict__ cf, float* __restrict__ cr,
    float* __restrict__ seqf, float* __restrict__ seqr) {
  const int H = 256, ZLD = 1024, SEQLD = 512;
  int lane = threadIdx.x & 31, wave = threadIdx.x >> 5;
  int half = lane >> 4, r = lane & 15;
  int dir = blockIdx.y;
  const float*  zx  = dir ? zxr   : zxf;
  const bf16_t* Whh = dir ? Whr   : Whf;
  const bf16_t* hin = dir ? hr_in : hf_in;
  bf16_t* hout = dir ? hr_out : hf_out;
  float*  c    = dir ? cr     : cf;
  float*  seq  = dir ? seqr   : seqf;
  int col = blockIdx.x * 64 + wave * 16 + r;  // hidden unit 0..255

  v8f acc[2][4];
#pragma unroll
  for (int mt = 0; mt < 2; ++mt)
#pragma unroll
    for (int g = 0; g < 4; ++g) acc[mt][g] = v8f_zero();

  const bf16_t* a0 = hin + (size_t)r * H;
  const bf16_t* a1 = hin + (size_t)(16 + r) * H;
  const bf16_t* bg[4];
#pragma unroll
  for (int g = 0; g < 4; ++g) bg[g] = Whh + (size_t)(g * H + col) * H;

  for (int k0 = 0; k0 < H; k0 += 32) {
    v16bf af0 = frag_a_bf16(a0, k0, half);
    v16bf af1 = frag_a_bf16(a1, k0, half);
#pragma unroll
    for (int g = 0; g < 4; ++g) {
      v16bf bfr = frag_b_bf16(bg[g], k0, half);
      acc[0][g] = wmma_bf16(af0, bfr, acc[0][g]);
      acc[1][g] = wmma_bf16(af1, bfr, acc[1][g]);
    }
  }
#pragma unroll
  for (int mt = 0; mt < 2; ++mt) {
#pragma unroll
    for (int v = 0; v < 8; ++v) {
      int m = mt * 16 + half * 8 + v;
      size_t zb = (size_t)m * ZLD;
      float zi = acc[mt][0][v] + zx[zb + 0 * H + col];
      float zf = acc[mt][1][v] + zx[zb + 1 * H + col];
      float zg = acc[mt][2][v] + zx[zb + 2 * H + col];
      float zo = acc[mt][3][v] + zx[zb + 3 * H + col];
      size_t si = (size_t)m * H + col;
      float cn = sigm_f(zf) * c[si] + sigm_f(zi) * tanh_f(zg);
      float hn = sigm_f(zo) * tanh_f(cn);
      c[si] = cn;
      hout[si] = (__bf16)hn;
      seq[(size_t)m * SEQLD + col] = hn;
    }
  }
}

// ---------------------------------------------------------------------------
// Decoder LSTM cell: z = x*Wih^T + h*Whh^T + b ; grid = {H/64}, block 128.
// x and h are bf16; c f32 in-place; h output bf16.
// ---------------------------------------------------------------------------
__global__ __launch_bounds__(128) void lstm_cell(
    const bf16_t* __restrict__ x, int Kx,
    const bf16_t* __restrict__ hin, int H,
    const bf16_t* __restrict__ Wih, const bf16_t* __restrict__ Whh,
    const float* __restrict__ bias,
    float* __restrict__ c, bf16_t* __restrict__ hout) {
  int lane = threadIdx.x & 31, wave = threadIdx.x >> 5;
  int half = lane >> 4, r = lane & 15;
  int col = blockIdx.x * 64 + wave * 16 + r;  // 0..H-1
  v8f acc[2][4];
#pragma unroll
  for (int mt = 0; mt < 2; ++mt)
#pragma unroll
    for (int g = 0; g < 4; ++g) acc[mt][g] = v8f_zero();

  {  // phase 1: x @ Wih^T
    const bf16_t* a0 = x + (size_t)r * Kx;
    const bf16_t* a1 = x + (size_t)(16 + r) * Kx;
    const bf16_t* bg[4];
#pragma unroll
    for (int g = 0; g < 4; ++g) bg[g] = Wih + (size_t)(g * H + col) * Kx;
    for (int k0 = 0; k0 < Kx; k0 += 32) {
      v16bf af0 = frag_a_bf16(a0, k0, half);
      v16bf af1 = frag_a_bf16(a1, k0, half);
#pragma unroll
      for (int g = 0; g < 4; ++g) {
        v16bf bfr = frag_b_bf16(bg[g], k0, half);
        acc[0][g] = wmma_bf16(af0, bfr, acc[0][g]);
        acc[1][g] = wmma_bf16(af1, bfr, acc[1][g]);
      }
    }
  }
  {  // phase 2: h @ Whh^T
    const bf16_t* a0 = hin + (size_t)r * H;
    const bf16_t* a1 = hin + (size_t)(16 + r) * H;
    const bf16_t* bg[4];
#pragma unroll
    for (int g = 0; g < 4; ++g) bg[g] = Whh + (size_t)(g * H + col) * H;
    for (int k0 = 0; k0 < H; k0 += 32) {
      v16bf af0 = frag_a_bf16(a0, k0, half);
      v16bf af1 = frag_a_bf16(a1, k0, half);
#pragma unroll
      for (int g = 0; g < 4; ++g) {
        v16bf bfr = frag_b_bf16(bg[g], k0, half);
        acc[0][g] = wmma_bf16(af0, bfr, acc[0][g]);
        acc[1][g] = wmma_bf16(af1, bfr, acc[1][g]);
      }
    }
  }
#pragma unroll
  for (int mt = 0; mt < 2; ++mt) {
#pragma unroll
    for (int v = 0; v < 8; ++v) {
      int m = mt * 16 + half * 8 + v;
      float zi = acc[mt][0][v] + bias[0 * H + col];
      float zf = acc[mt][1][v] + bias[1 * H + col];
      float zg = acc[mt][2][v] + bias[2 * H + col];
      float zo = acc[mt][3][v] + bias[3 * H + col];
      size_t si = (size_t)m * H + col;
      float cn = sigm_f(zf) * c[si] + sigm_f(zi) * tanh_f(zg);
      float hn = sigm_f(zo) * tanh_f(cn);
      c[si] = cn;
      hout[si] = (__bf16)hn;
    }
  }
}

// ---------------------------------------------------------------------------
// Small helper kernels
// ---------------------------------------------------------------------------
__global__ void f32_to_bf16_k(const float* __restrict__ in, bf16_t* __restrict__ out,
                              size_t n) {
  size_t i = (size_t)blockIdx.x * blockDim.x + threadIdx.x;
  if (i < n) out[i] = (__bf16)in[i];
}

__global__ void f32_to_bf16_pad(const float* __restrict__ src, bf16_t* __restrict__ dst,
                                int rows, int Ks, int Kd) {
  int i = blockIdx.x * blockDim.x + threadIdx.x;
  if (i >= rows * Kd) return;
  int row = i / Kd, col = i % Kd;
  float z = 0.0f;
  dst[i] = (col < Ks) ? (__bf16)src[(size_t)row * Ks + col] : (__bf16)z;
}

__global__ void zero_bf16(bf16_t* __restrict__ p, size_t n) {
  size_t i = (size_t)blockIdx.x * blockDim.x + threadIdx.x;
  float z = 0.0f;
  if (i < n) p[i] = (__bf16)z;
}

__global__ void zero_f32(float* __restrict__ p, size_t n) {
  size_t i = (size_t)blockIdx.x * blockDim.x + threadIdx.x;
  if (i < n) p[i] = 0.0f;
}

// dst[row][0:Kd] = (col < Ks) ? src[row][col] : 0   (f32 K-padding copy)
__global__ void pad_copy_f32(const float* __restrict__ src, float* __restrict__ dst,
                             size_t rows, int Ks, int Kd) {
  size_t i = (size_t)blockIdx.x * blockDim.x + threadIdx.x;
  if (i >= rows * (size_t)Kd) return;
  int col = (int)(i % Kd);
  size_t row = i / Kd;
  dst[i] = (col < Ks) ? src[row * Ks + col] : 0.0f;
}

// Y[t2][b][:] = X[2t2][b][:] ++ X[2t2+1][b][:]
__global__ void pack_stride2(const float* __restrict__ X, float* __restrict__ Y,
                             int T2, int B, int D) {
  size_t i = (size_t)blockIdx.x * blockDim.x + threadIdx.x;
  size_t total = (size_t)T2 * B * 2 * D;
  if (i >= total) return;
  int d = (int)(i % (2 * D));
  size_t mb = i / (2 * D);
  int b = (int)(mb % B);
  int t2 = (int)(mb / B);
  int toff = d / D, dd = d % D;
  Y[i] = X[((size_t)(2 * t2 + toff) * B + b) * D + dd];
}

__global__ void concat2_bf16(const bf16_t* __restrict__ a, const bf16_t* __restrict__ b,
                             bf16_t* __restrict__ o, int rows, int D) {
  int i = blockIdx.x * blockDim.x + threadIdx.x;
  int tot = rows * 2 * D;
  if (i >= tot) return;
  int r = i / (2 * D), d = i % (2 * D);
  o[i] = (d < D) ? a[(size_t)r * D + d] : b[(size_t)r * D + (d - D)];
}

__global__ void attn_energy(const float* __restrict__ q, const float* __restrict__ Kt,
                            float* __restrict__ energy, int S, int B, int D) {
  int b = blockIdx.x, s = threadIdx.x;
  if (s >= S) return;
  const float* qb = q + (size_t)b * D;
  const float* kb = Kt + ((size_t)s * B + b) * D;
  float sum = 0.0f;
  for (int d = 0; d < D; d += 4) {
    float4 qv = *(const float4*)(qb + d);
    float4 kv = *(const float4*)(kb + d);
    sum += qv.x * kv.x + qv.y * kv.y + qv.z * kv.z + qv.w * kv.w;
  }
  energy[(size_t)b * S + s] = sum;
}

__global__ void attn_softmax(float* __restrict__ e, int S) {
  __shared__ float red[128];
  int b = blockIdx.x, s = threadIdx.x;
  float v = e[(size_t)b * S + s];
  red[s] = v;
  __syncthreads();
  for (int st = 64; st >= 1; st >>= 1) {
    if (s < st) red[s] = fmaxf(red[s], red[s + st]);
    __syncthreads();
  }
  float mx = red[0];
  __syncthreads();
  float ex = __expf(v - mx);
  red[s] = ex;
  __syncthreads();
  for (int st = 64; st >= 1; st >>= 1) {
    if (s < st) red[s] += red[s + st];
    __syncthreads();
  }
  float sum = red[0];
  e[(size_t)b * S + s] = ex / fmaxf(sum, 1e-12f);
}

__global__ void attn_ctx(const float* __restrict__ a, const float* __restrict__ V,
                         float* __restrict__ ctx, int S, int B, int D) {
  int b = blockIdx.x;
  for (int d = threadIdx.x; d < D; d += blockDim.x) {
    float sum = 0.0f;
    for (int s = 0; s < S; ++s)
      sum += a[(size_t)b * S + s] * V[((size_t)s * B + b) * D + d];
    ctx[(size_t)b * D + d] = sum;
  }
}

// xx[b][0:512] = embed[word[b]] ; xx[b][512:1024] = ctx[b]  (emit bf16)
__global__ void build_xx_bf16(const int* __restrict__ words_t,
                              const float* __restrict__ embed,
                              const float* __restrict__ ctx,
                              bf16_t* __restrict__ xx) {
  int i = blockIdx.x * blockDim.x + threadIdx.x;
  if (i >= 32 * 1024) return;
  int b = i >> 10, d = i & 1023;
  float v;
  if (d < 512) v = embed[(size_t)words_t[b] * 512 + d];
  else         v = ctx[(size_t)b * 512 + (d - 512)];
  xx[i] = (__bf16)v;
}

// ---------------------------------------------------------------------------
// Host orchestration
// ---------------------------------------------------------------------------
extern "C" void kernel_launch(void* const* d_in, const int* in_sizes, int n_in,
                              void* d_out, int out_size, void* d_ws, size_t ws_size,
                              hipStream_t stream) {
  (void)in_sizes; (void)n_in; (void)out_size; (void)ws_size;
  const float* inputs = (const float*)d_in[0];       // (1024, 32, 40)
  const int*   words  = (const int*)d_in[1];         // (200, 32)
  const float* encW[8][3];
  int pi = 2;
  for (int i = 0; i < 8; ++i)
    for (int j = 0; j < 3; ++j) encW[i][j] = (const float*)d_in[pi++];
  const float* fc1w = (const float*)d_in[26];
  const float* fc1b = (const float*)d_in[27];
  const float* fc2w = (const float*)d_in[28];
  const float* fc2b = (const float*)d_in[29];
  const float* qw   = (const float*)d_in[30];
  const float* qb   = (const float*)d_in[31];
  const float* embed = (const float*)d_in[32];       // (5000, 512)
  const float* d1Wih = (const float*)d_in[33];       // (2048, 1024)
  const float* d1Whh = (const float*)d_in[34];       // (2048, 512)
  const float* d1b   = (const float*)d_in[35];
  const float* d2Wih = (const float*)d_in[36];       // (2048, 512)
  const float* d2Whh = (const float*)d_in[37];       // (2048, 512)
  const float* d2b   = (const float*)d_in[38];
  const float* fcb   = (const float*)d_in[39];       // (5000,)
  float* out = (float*)d_out;                        // (200, 32, 5000)

  // -------- workspace bump allocator --------
  char* wsp = (char*)d_ws;
  auto alloc = [&](size_t bytes) -> void* {
    void* p = (void*)wsp;
    wsp += (bytes + 255) & ~(size_t)255;
    return p;
  };
  auto conv = [&](const float* src, bf16_t* dst, size_t n) {
    f32_to_bf16_k<<<(unsigned)((n + 255) / 256), 256, 0, stream>>>(src, dst, n);
  };
  auto zero = [&](float* p, size_t n) {
    zero_f32<<<(unsigned)((n + 255) / 256), 256, 0, stream>>>(p, n);
  };
  auto zerob = [&](bf16_t* p, size_t n) {
    zero_bf16<<<(unsigned)((n + 255) / 256), 256, 0, stream>>>(p, n);
  };
  auto gemm = [&](const float* A, const bf16_t* B, const float* bias, float* C,
                  int M, int N, int K, int act) {
    dim3 g((N + 127) / 128, (M + 15) / 16);
    gemm_bias_act<<<g, 256, 0, stream>>>(A, B, bias, C, M, N, K, act);
  };
  auto gemmb = [&](const bf16_t* A, const bf16_t* B, const float* bias, float* C,
                   int M, int N, int K, int act) {
    dim3 g((N + 127) / 128, (M + 15) / 16);
    gemm_bf16A<<<g, 256, 0, stream>>>(A, B, bias, C, M, N, K, act);
  };

  // -------- bf16 weight buffers + conversion --------
  bf16_t* encWih_bf[8];
  bf16_t* encWhh_bf[8];
  for (int i = 0; i < 8; ++i) {
    if (i < 2) {  // layer-1 Wih padded K 40 -> 64
      encWih_bf[i] = (bf16_t*)alloc((size_t)1024 * 64 * 2);
      f32_to_bf16_pad<<<(1024 * 64 + 255) / 256, 256, 0, stream>>>(
          encW[i][0], encWih_bf[i], 1024, 40, 64);
    } else {
      encWih_bf[i] = (bf16_t*)alloc((size_t)1024 * 1024 * 2);
      conv(encW[i][0], encWih_bf[i], (size_t)1024 * 1024);
    }
    encWhh_bf[i] = (bf16_t*)alloc((size_t)1024 * 256 * 2);
    conv(encW[i][1], encWhh_bf[i], (size_t)1024 * 256);
  }
  bf16_t* fc1w_bf = (bf16_t*)alloc((size_t)512 * 512 * 2);  conv(fc1w, fc1w_bf, 512 * 512);
  bf16_t* fc2w_bf = (bf16_t*)alloc((size_t)512 * 512 * 2);  conv(fc2w, fc2w_bf, 512 * 512);
  bf16_t* qw_bf   = (bf16_t*)alloc((size_t)512 * 512 * 2);  conv(qw, qw_bf, 512 * 512);
  // embedding-as-logits-weight padded N 5000 -> 5120 rows (zero tail)
  bf16_t* emb_bf = (bf16_t*)alloc((size_t)5120 * 512 * 2);
  conv(embed, emb_bf, (size_t)5000 * 512);
  zerob(emb_bf + (size_t)5000 * 512, (size_t)120 * 512);
  bf16_t* d1Wih_bf = (bf16_t*)alloc((size_t)2048 * 1024 * 2); conv(d1Wih, d1Wih_bf, (size_t)2048 * 1024);
  bf16_t* d1Whh_bf = (bf16_t*)alloc((size_t)2048 * 512 * 2);  conv(d1Whh, d1Whh_bf, (size_t)2048 * 512);
  bf16_t* d2Wih_bf = (bf16_t*)alloc((size_t)2048 * 512 * 2);  conv(d2Wih, d2Wih_bf, (size_t)2048 * 512);
  bf16_t* d2Whh_bf = (bf16_t*)alloc((size_t)2048 * 512 * 2);  conv(d2Whh, d2Whh_bf, (size_t)2048 * 512);

  // -------- activation buffers --------
  float* in_pad = (float*)alloc((size_t)1024 * 32 * 64 * 4);  // K 40 -> 64
  pad_copy_f32<<<(unsigned)(((size_t)1024 * 32 * 64 + 255) / 256), 256, 0, stream>>>(
      inputs, in_pad, (size_t)1024 * 32, 40, 64);
  float* zxf = (float*)alloc((size_t)1024 * 32 * 1024 * 4);
  float* zxr = (float*)alloc((size_t)1024 * 32 * 1024 * 4);
  float* X[4];
  float* Y[3];
  X[0] = (float*)alloc((size_t)1024 * 32 * 512 * 4);
  Y[0] = (float*)alloc((size_t)512 * 32 * 1024 * 4);
  X[1] = (float*)alloc((size_t)512 * 32 * 512 * 4);
  Y[1] = (float*)alloc((size_t)256 * 32 * 1024 * 4);
  X[2] = (float*)alloc((size_t)256 * 32 * 512 * 4);
  Y[2] = (float*)alloc((size_t)128 * 32 * 1024 * 4);
  X[3] = (float*)alloc((size_t)128 * 32 * 512 * 4);
  float* Kbuf = (float*)alloc((size_t)4096 * 512 * 4);
  float* Vbuf = (float*)alloc((size_t)4096 * 512 * 4);
  // encoder h state (bf16 ping-pong) + c state (f32)
  bf16_t* hfb[2] = {(bf16_t*)alloc(32 * 256 * 2), (bf16_t*)alloc(32 * 256 * 2)};
  bf16_t* hrb[2] = {(bf16_t*)alloc(32 * 256 * 2), (bf16_t*)alloc(32 * 256 * 2)};
  float* cfb = (float*)alloc(32 * 256 * 4);
  float* crb = (float*)alloc(32 * 256 * 4);
  bf16_t* henc_bf = (bf16_t*)alloc(32 * 512 * 2);
  float* qbuf   = (float*)alloc(32 * 512 * 4);
  float* energy = (float*)alloc(32 * 128 * 4);
  float* ctx    = (float*)alloc(32 * 512 * 4);
  bf16_t* xx_bf = (bf16_t*)alloc(32 * 1024 * 2);
  bf16_t* h1p[2] = {(bf16_t*)alloc(32 * 512 * 2), (bf16_t*)alloc(32 * 512 * 2)};
  bf16_t* h2p[2] = {(bf16_t*)alloc(32 * 512 * 2), (bf16_t*)alloc(32 * 512 * 2)};
  float* c1 = (float*)alloc(32 * 512 * 4);
  float* c2 = (float*)alloc(32 * 512 * 4);

  // -------- encoder: 4 pyramidal bi-LSTM layers --------
  const float* cur = in_pad;
  int T = 1024, Kin = 64;
  const bf16_t* hf_final = nullptr;
  const bf16_t* hr_final = nullptr;
  for (int l = 0; l < 4; ++l) {
    int M = T * 32;
    // big parallel GEMMs: zx = x @ Wih^T + b for all timesteps, both dirs
    gemm(cur, encWih_bf[2 * l],     encW[2 * l][2],     zxf, M, 1024, Kin, ACT_NONE);
    gemm(cur, encWih_bf[2 * l + 1], encW[2 * l + 1][2], zxr, M, 1024, Kin, ACT_NONE);
    zerob(hfb[0], 32 * 256); zerob(hrb[0], 32 * 256);
    zero(cfb, 32 * 256);     zero(crb, 32 * 256);
    for (int s = 0; s < T; ++s) {
      int tf = s, tr = T - 1 - s;
      lstm_step_enc<<<dim3(4, 2), 128, 0, stream>>>(
          zxf + (size_t)tf * 32 * 1024, zxr + (size_t)tr * 32 * 1024,
          encWhh_bf[2 * l], encWhh_bf[2 * l + 1],
          hfb[s & 1], hrb[s & 1], hfb[(s + 1) & 1], hrb[(s + 1) & 1],
          cfb, crb,
          X[l] + (size_t)tf * 32 * 512,
          X[l] + (size_t)tr * 32 * 512 + 256);
    }
    if (l < 3) {
      size_t tot = (size_t)(T / 2) * 32 * 1024;
      pack_stride2<<<(unsigned)((tot + 255) / 256), 256, 0, stream>>>(X[l], Y[l], T / 2, 32, 512);
      cur = Y[l];
      T /= 2;
      Kin = 1024;
    } else {
      hf_final = hfb[T & 1];
      hr_final = hrb[T & 1];
    }
  }

  // -------- K, V projections + encoder summary --------
  gemm(X[3], fc1w_bf, fc1b, Kbuf, 128 * 32, 512, 512, ACT_SELU);
  gemm(X[3], fc2w_bf, fc2b, Vbuf, 128 * 32, 512, 512, ACT_SELU);
  concat2_bf16<<<(32 * 512 + 255) / 256, 256, 0, stream>>>(hf_final, hr_final,
                                                           henc_bf, 32, 256);

  // -------- decoder: 199 attention + 2-cell steps --------
  zerob(h1p[0], 32 * 512); zero(c1, 32 * 512);
  zerob(h2p[0], 32 * 512); zero(c2, 32 * 512);
  zero(out, (size_t)32 * 5000);  // pred0 = zeros
  const bf16_t* qsrc = henc_bf;  // first step attends with h_enc
  for (int i = 0; i < 199; ++i) {
    gemmb(qsrc, qw_bf, qb, qbuf, 32, 512, 512, ACT_SELU);
    attn_energy<<<32, 128, 0, stream>>>(qbuf, Kbuf, energy, 128, 32, 512);
    attn_softmax<<<32, 128, 0, stream>>>(energy, 128);
    attn_ctx<<<32, 256, 0, stream>>>(energy, Vbuf, ctx, 128, 32, 512);
    build_xx_bf16<<<(32 * 1024 + 255) / 256, 256, 0, stream>>>(words + i * 32, embed,
                                                               ctx, xx_bf);
    lstm_cell<<<8, 128, 0, stream>>>(xx_bf, 1024, h1p[i & 1], 512,
                                     d1Wih_bf, d1Whh_bf, d1b, c1, h1p[(i + 1) & 1]);
    lstm_cell<<<8, 128, 0, stream>>>(h1p[(i + 1) & 1], 512, h2p[i & 1], 512,
                                     d2Wih_bf, d2Whh_bf, d2b, c2, h2p[(i + 1) & 1]);
    gemmb(h2p[(i + 1) & 1], emb_bf, fcb, out + (size_t)(i + 1) * 32 * 5000,
          32, 5000, 512, ACT_NONE);
    qsrc = h2p[(i + 1) & 1];
  }
}
